// RSNNCell_12575664243445
// MI455X (gfx1250) — matrix operand order repaired and checked by reading, not compiled, not run on hardware
//
#include <hip/hip_runtime.h>

// ---------------------------------------------------------------------------
// RSNN / LIF cell step for MI455X (gfx1250, wave32, WMMA).
//
// GEMM1: i_t = [inputs|z](f16 spikes, exact) @ [[w_in];[w_rec]](f16)   K=4096
//        wave tile 32x64 (8 accumulators) -> 8 wmma per 12 b128-frag loads
// fused LIF epilogue -> new_v, new_z, new_r, dz_dh (+ new_z repacked f16)
// GEMM2: new_out = KAPPA*out + new_z(f16) @ w_out(f16)                 K=2048
//        wave tile 16x32 (1024 waves for occupancy on the small tail GEMM)
// traces: elementwise, float4 streams.
//
// All f16 operands (30 MB) fit in the 192 MB L2, so fragments are streamed
// directly global->VGPR (2x b128 per fragment per lane); no LDS staging.
// ---------------------------------------------------------------------------

typedef __attribute__((ext_vector_type(16))) _Float16 v16h;
typedef __attribute__((ext_vector_type(8)))  float    v8f;

constexpr int Bm   = 1024;   // batch
constexpr int NIN  = 2048;
constexpr int Hn   = 2048;
constexpr int NOUT = 512;
constexpr int K1   = NIN + Hn;   // 4096 (concatenated GEMM1 K)

constexpr float kDECAY = 0.95122942450071400909f;  // exp(-1/20)
constexpr float kKAPPA = 0.95122942450071400909f;
constexpr float kTHR   = 0.4f;
constexpr float kINVT  = 2.5f;                     // 1/THR
constexpr float kNREF  = 5.0f;
constexpr float kDAMP  = 0.3f;

// ---------------------------------------------------------------------------
// Fragment packing.
// A-fragment (16x32 f16, ISA 7.12.2): lane<16: M=lane, K in {0..7,16..23};
//   lane>=16: M=lane-16, K in {8..15,24..31}; half h: K=(h&7)+8*(lane>=16)+16*(h>>3)
// B-fragment (32x16 f16, same K order as A): lane<16: N=lane, K in {0..7,16..23};
//   lane>=16: N=lane-16, K in {8..15,24..31}; half h: K as for A.
// Storage: fragment-major, frag stride = 512 halves, per-lane 16 contiguous
// halves => each lane loads its fragment as two b128s.
// ---------------------------------------------------------------------------

__global__ void pack_A_concat(const float* __restrict__ X0, const float* __restrict__ X1,
                              _Float16* __restrict__ dst, int K, int K0)
{
    // one thread per destination half; writes fully coalesced, reads via L2
    size_t t   = (size_t)blockIdx.x * blockDim.x + threadIdx.x;
    int    KT  = K >> 5;
    size_t fr  = t >> 9;
    int    idx = (int)(t & 511);
    int    lane = idx >> 4, h = idx & 15;
    int    mt = (int)(fr / KT), kt = (int)(fr % KT);
    int    m  = mt * 16 + (lane & 15);
    int    hi = lane >> 4;
    int    k  = kt * 32 + (h & 7) + 8 * hi + ((h >> 3) << 4);
    float  val = (k < K0) ? X0[(size_t)m * K0 + k]
                          : X1[(size_t)m * (K - K0) + (k - K0)];
    dst[t] = (_Float16)val;
}

__global__ void pack_B_concat(const float* __restrict__ W0, const float* __restrict__ W1,
                              _Float16* __restrict__ dst, int K, int N, int K0)
{
    size_t t   = (size_t)blockIdx.x * blockDim.x + threadIdx.x;
    int    KT  = K >> 5;
    size_t fr  = t >> 9;
    int    idx = (int)(t & 511);
    int    lane = idx >> 4, h = idx & 15;
    int    nt = (int)(fr / KT), kt = (int)(fr % KT);
    int    n  = nt * 16 + (lane & 15);
    int    k  = kt * 32 + (h & 7) + 8 * (lane >> 4) + ((h >> 3) << 4);
    float  val = (k < K0) ? W0[(size_t)k * N + n]
                          : W1[(size_t)(k - K0) * N + n];
    dst[t] = (_Float16)val;
}

// ---------------------------------------------------------------------------
// C/D layout helpers (16x16 f32 acc: lane<16 -> M=g,N=lane; lane>=16 -> M=g+8)
// ---------------------------------------------------------------------------
__device__ __forceinline__ v8f load_c_scaled(const float* __restrict__ src,
                                             int R0, int C0, int lane, int ld, float s)
{
    v8f c;
    const int hi  = lane >> 4;
    const int col = C0 + (lane & 15);
#pragma unroll
    for (int g = 0; g < 8; ++g)
        c[g] = s * src[(size_t)(R0 + g + 8 * hi) * ld + col];
    return c;
}

__device__ __forceinline__ void store_c(float* __restrict__ dst, v8f c,
                                        int R0, int C0, int lane, int ld)
{
    const int hi  = lane >> 4;
    const int col = C0 + (lane & 15);
#pragma unroll
    for (int g = 0; g < 8; ++g)
        dst[(size_t)(R0 + g + 8 * hi) * ld + col] = c[g];
}

// Fused LIF epilogue for one 16x16 accumulator tile of i_t.
__device__ __forceinline__ void lif_epilogue_tile(
    v8f c, int R0, int C0, int lane,
    const float* __restrict__ vin, const float* __restrict__ zin,
    const float* __restrict__ rin,
    float* __restrict__ oz, float* __restrict__ ov,
    float* __restrict__ orr, float* __restrict__ odz,
    _Float16* __restrict__ zfrag)
{
    const int hi  = lane >> 4;
    const int col = C0 + (lane & 15);
#pragma unroll
    for (int g = 0; g < 8; ++g) {
        const int    row = R0 + g + 8 * hi;
        const size_t idx = (size_t)row * Hn + col;
        const float  i_t  = c[g];
        const float  vold = vin[idx];
        const float  zold = zin[idx];
        const float  rold = rin[idx];
        const float  nv   = kDECAY * vold + i_t - zold * kTHR;
        const float  vs   = (nv - kTHR) * kINVT;
        const bool   refr = rold > 0.1f;
        float nz = (vs > 0.0f) ? 1.0f : 0.0f;
        if (refr) nz = 0.0f;
        const float nr = fminf(fmaxf(rold + kNREF * nz - 1.0f, 0.0f), kNREF);
        float dz = fmaxf(kDAMP * (1.0f - fabsf(vs)), 0.0f) * kINVT;
        if (refr) dz = 0.0f;
        ov[idx]  = nv;
        oz[idx]  = nz;
        orr[idx] = nr;
        odz[idx] = dz;
        // scatter new_z (f16) straight into A-fragment layout for GEMM2
        const int mt = row >> 4, ml = row & 15;
        const int kt = col >> 5, kl = col & 31;
        const int la = ml + (((kl >> 3) & 1) << 4);
        const int hh = (kl & 7) + ((kl >> 4) << 3);
        zfrag[(((size_t)mt * (Hn / 32) + kt) * 32 + la) * 16 + hh] = (_Float16)nz;
    }
}

#define WMMA_F16(A, B, C) \
    __builtin_amdgcn_wmma_f32_16x16x32_f16(false, (A), false, (B), (short)0, (C), false, false)

// ---------------------------------------------------------------------------
// GEMM1: i_t = A(1024x4096) @ B(4096x2048), wave = 32x64 C tile, fused LIF.
// 8 wmma per 6 fragment (12 b128) loads => 21.3 FLOP per L2 byte.
// ---------------------------------------------------------------------------
__global__ void __launch_bounds__(256) lif_gemm1(
    const _Float16* __restrict__ Af, const _Float16* __restrict__ Bf,
    const float* __restrict__ vin, const float* __restrict__ zin,
    const float* __restrict__ rin,
    float* __restrict__ oz, float* __restrict__ ov,
    float* __restrict__ orr, float* __restrict__ odz,
    _Float16* __restrict__ zfrag)
{
    constexpr int KT = K1 / 32;   // 128
    constexpr int TN = Hn / 64;   // 32 tile-cols (64-wide)
    const int wave = threadIdx.x >> 5;
    const int lane = threadIdx.x & 31;
    const int tile = blockIdx.x * 8 + wave;
    const int tm   = tile / TN;
    const int tn   = tile % TN;

    const _Float16* a0p = Af + ((size_t)(tm * 2)     * KT) * 512 + lane * 16;
    const _Float16* a1p = Af + ((size_t)(tm * 2 + 1) * KT) * 512 + lane * 16;
    const _Float16* b0p = Bf + ((size_t)(tn * 4)     * KT) * 512 + lane * 16;
    const _Float16* b1p = Bf + ((size_t)(tn * 4 + 1) * KT) * 512 + lane * 16;
    const _Float16* b2p = Bf + ((size_t)(tn * 4 + 2) * KT) * 512 + lane * 16;
    const _Float16* b3p = Bf + ((size_t)(tn * 4 + 3) * KT) * 512 + lane * 16;

    v8f c00 = {}, c01 = {}, c02 = {}, c03 = {};
    v8f c10 = {}, c11 = {}, c12 = {}, c13 = {};
#pragma unroll 2
    for (int kt = 0; kt < KT; ++kt) {
        v16h a0 = *(const v16h*)a0p; a0p += 512;
        v16h a1 = *(const v16h*)a1p; a1p += 512;
        v16h b0 = *(const v16h*)b0p; b0p += 512;
        v16h b1 = *(const v16h*)b1p; b1p += 512;
        v16h b2 = *(const v16h*)b2p; b2p += 512;
        v16h b3 = *(const v16h*)b3p; b3p += 512;
        c00 = WMMA_F16(a0, b0, c00);
        c01 = WMMA_F16(a0, b1, c01);
        c02 = WMMA_F16(a0, b2, c02);
        c03 = WMMA_F16(a0, b3, c03);
        c10 = WMMA_F16(a1, b0, c10);
        c11 = WMMA_F16(a1, b1, c11);
        c12 = WMMA_F16(a1, b2, c12);
        c13 = WMMA_F16(a1, b3, c13);
    }
    const int R0 = tm * 32, C0 = tn * 64;
    lif_epilogue_tile(c00, R0,      C0,      lane, vin, zin, rin, oz, ov, orr, odz, zfrag);
    lif_epilogue_tile(c01, R0,      C0 + 16, lane, vin, zin, rin, oz, ov, orr, odz, zfrag);
    lif_epilogue_tile(c02, R0,      C0 + 32, lane, vin, zin, rin, oz, ov, orr, odz, zfrag);
    lif_epilogue_tile(c03, R0,      C0 + 48, lane, vin, zin, rin, oz, ov, orr, odz, zfrag);
    lif_epilogue_tile(c10, R0 + 16, C0,      lane, vin, zin, rin, oz, ov, orr, odz, zfrag);
    lif_epilogue_tile(c11, R0 + 16, C0 + 16, lane, vin, zin, rin, oz, ov, orr, odz, zfrag);
    lif_epilogue_tile(c12, R0 + 16, C0 + 32, lane, vin, zin, rin, oz, ov, orr, odz, zfrag);
    lif_epilogue_tile(c13, R0 + 16, C0 + 48, lane, vin, zin, rin, oz, ov, orr, odz, zfrag);
}

// ---------------------------------------------------------------------------
// GEMM2: new_out = KAPPA*out + Z(1024x2048) @ Wout(2048x512)
// wave = 16x32 C tile -> 1024 waves (occupancy for the small tail GEMM).
// ---------------------------------------------------------------------------
__global__ void __launch_bounds__(256) lif_gemm2(
    const _Float16* __restrict__ Zf, const _Float16* __restrict__ Wf,
    const float* __restrict__ oldout, float* __restrict__ onew)
{
    constexpr int KT = Hn / 32;     // 64
    constexpr int TN = NOUT / 32;   // 16 tile-cols (32-wide)
    const int wave = threadIdx.x >> 5;
    const int lane = threadIdx.x & 31;
    const int tile = blockIdx.x * 8 + wave;
    const int tm   = tile / TN;     // 16-row tile
    const int tn   = tile % TN;
    const int R0 = tm * 16, C0 = tn * 32;

    const _Float16* a0p = Zf + ((size_t)tm * KT) * 512 + lane * 16;
    const _Float16* b0p = Wf + ((size_t)(tn * 2)     * KT) * 512 + lane * 16;
    const _Float16* b1p = Wf + ((size_t)(tn * 2 + 1) * KT) * 512 + lane * 16;

    v8f c00 = load_c_scaled(oldout, R0, C0,      lane, NOUT, kKAPPA);
    v8f c01 = load_c_scaled(oldout, R0, C0 + 16, lane, NOUT, kKAPPA);
#pragma unroll 2
    for (int kt = 0; kt < KT; ++kt) {
        v16h a0 = *(const v16h*)a0p; a0p += 512;
        v16h b0 = *(const v16h*)b0p; b0p += 512;
        v16h b1 = *(const v16h*)b1p; b1p += 512;
        c00 = WMMA_F16(a0, b0, c00);
        c01 = WMMA_F16(a0, b1, c01);
    }
    store_c(onew, c00, R0, C0,      lane, NOUT);
    store_c(onew, c01, R0, C0 + 16, lane, NOUT);
}

// ---------------------------------------------------------------------------
// traces: new_trace_in = trace_in*DECAY + inputs ; same for rec with z
// ---------------------------------------------------------------------------
__global__ void lif_traces(const float4* __restrict__ tin,  const float4* __restrict__ inp,
                           const float4* __restrict__ trec, const float4* __restrict__ z,
                           float4* __restrict__ otin, float4* __restrict__ otrec)
{
    const size_t i = (size_t)blockIdx.x * blockDim.x + threadIdx.x;
    float4 a = tin[i],  b = inp[i];
    float4 c = trec[i], d = z[i];
    otin[i]  = make_float4(a.x * kDECAY + b.x, a.y * kDECAY + b.y,
                           a.z * kDECAY + b.z, a.w * kDECAY + b.w);
    otrec[i] = make_float4(c.x * kDECAY + d.x, c.y * kDECAY + d.y,
                           c.z * kDECAY + d.z, c.w * kDECAY + d.w);
}

// ---------------------------------------------------------------------------
extern "C" void kernel_launch(void* const* d_in, const int* in_sizes, int n_in,
                              void* d_out, int out_size, void* d_ws, size_t ws_size,
                              hipStream_t stream)
{
    const float* inputs = (const float*)d_in[0];
    const float* v      = (const float*)d_in[1];
    const float* z      = (const float*)d_in[2];
    const float* r      = (const float*)d_in[3];
    const float* out0   = (const float*)d_in[4];
    const float* tin    = (const float*)d_in[5];
    const float* trec   = (const float*)d_in[6];
    const float* w_in   = (const float*)d_in[7];
    const float* w_rec  = (const float*)d_in[8];
    const float* w_out  = (const float*)d_in[9];

    float* out = (float*)d_out;
    // output tuple, flat concat in return order
    float* o_out  = out;                                   // [1024,512]
    float* o_z    = o_out  + (size_t)Bm * NOUT;            // [1024,2048]
    float* o_v    = o_z    + (size_t)Bm * Hn;
    float* o_r    = o_v    + (size_t)Bm * Hn;
    float* o_tin  = o_r    + (size_t)Bm * Hn;
    float* o_trec = o_tin  + (size_t)Bm * NIN;
    float* o_dz   = o_trec + (size_t)Bm * Hn;

    // workspace: A1 8MiB | B1 16MiB | Wout 2MiB | Zfrag 4MiB  (30 MiB total)
    char* ws = (char*)d_ws;
    _Float16* Af  = (_Float16*)(ws);
    _Float16* Bf  = (_Float16*)(ws + (8ull  << 20));
    _Float16* Wof = (_Float16*)(ws + (24ull << 20));
    _Float16* Zf  = (_Float16*)(ws + (26ull << 20));

    // 1) pack f32 -> f16 WMMA fragments
    pack_A_concat<<<(Bm * K1) / 256, 256, 0, stream>>>(inputs, z, Af, K1, NIN);
    pack_B_concat<<<(K1 * Hn) / 256, 256, 0, stream>>>(w_in, w_rec, Bf, K1, Hn, NIN);
    pack_B_concat<<<(Hn * NOUT) / 256, 256, 0, stream>>>(w_out, w_out, Wof, Hn, NOUT, Hn);

    // 2) recurrent GEMM + fused LIF update (writes z/v/r/dz + Zf f16 fragments)
    lif_gemm1<<<(Bm / 32) * (Hn / 64) / 8, 256, 0, stream>>>(
        Af, Bf, v, z, r, o_z, o_v, o_r, o_dz, Zf);

    // 3) output GEMM, accumulator seeded with KAPPA*out
    lif_gemm2<<<(Bm / 16) * (NOUT / 32) / 8, 256, 0, stream>>>(Zf, Wof, out0, o_out);

    // 4) eligibility traces
    lif_traces<<<(Bm * Hn / 4) / 256, 256, 0, stream>>>(
        (const float4*)tin, (const float4*)inputs,
        (const float4*)trec, (const float4*)z,
        (float4*)o_tin, (float4*)o_trec);
}
